// ScaledDotProductAttention_84782654423626
// MI455X (gfx1250) — matrix operand order repaired and checked by reading, not compiled, not run on hardware
//
#include <hip/hip_runtime.h>

typedef __bf16 bf16;
typedef __attribute__((ext_vector_type(16))) __bf16 v16bf;
typedef __attribute__((ext_vector_type(8)))  float  v8f;

#define WMMA_BF16(a, b, c) \
  __builtin_amdgcn_wmma_f32_16x16x32_bf16(false, (a), false, (b), (short)0, (c), false, false)

constexpr int   Bc = 2, Hc = 16, Sc = 2048, Dc = 64;
constexpr int   LDSW = 2052;                 // padded LDS row stride (floats): 2052 % 64 == 4 -> bank spread
constexpr float SCALE = 0.125f;              // 1/sqrt(64)
constexpr float NEGV  = -1.0e9f;

__global__ __launch_bounds__(128)
void sdpa_fwd(const float* __restrict__ Q, const float* __restrict__ K,
              const float* __restrict__ V, const int* __restrict__ AM,
              float* __restrict__ O, float* __restrict__ P)
{
    extern __shared__ float smem[];
    float* sc   = smem;                  // [16][LDSW] scores -> probabilities
    float* red  = smem + 16 * LDSW;      // [16*8] partial reductions
    float* mxA  = red + 128;             // [16] row max
    float* invA = mxA + 16;              // [16] 1/row-sum

    const int tid  = threadIdx.x;
    const int wave = tid >> 5;
    const int lane = tid & 31;
    const int lh   = lane >> 4;          // lane half (0/1)
    const int ln   = lane & 15;

    const int blk = blockIdx.x;
    const int qt  = blk & ((Sc / 16) - 1);   // q tile within head
    const int bh  = blk >> 7;                // b*H + h
    const int b   = bh >> 4;                 // H == 16
    const int q0  = qt << 4;

    const size_t headOff = (size_t)bh * Sc * Dc;
    const float* Qh = Q + headOff;
    const float* Kh = K + headOff;
    const float* Vh = V + headOff;
    const int*   Mb = AM + (size_t)b * Sc;

    // ---------------- Q tile -> two bf16 A fragments (K = 0..31, 32..63) ----------------
    // A layout (16-bit, 16x32): lanes 0-15 hold K {0..7, 16..23}; lanes 16-31 hold K {8..15, 24..31}
    // 1/sqrt(d) is folded into Q here (f32 multiply before bf16 rounding).
    v16bf aq[2];
    {
        const float* qrow = Qh + (size_t)(q0 + ln) * Dc;
        #pragma unroll
        for (int c = 0; c < 2; ++c) {
            const int s1 = c * 32 + (lh ? 8 : 0);
            float t[16];
            *(float4*)(t + 0)  = *(const float4*)(qrow + s1 + 0);
            *(float4*)(t + 4)  = *(const float4*)(qrow + s1 + 4);
            *(float4*)(t + 8)  = *(const float4*)(qrow + s1 + 16);
            *(float4*)(t + 12) = *(const float4*)(qrow + s1 + 20);
            v16bf f;
            #pragma unroll
            for (int e = 0; e < 16; ++e) f[e] = (bf16)(t[e] * SCALE);
            aq[c] = f;
        }
    }

    // ---------------- Phase A: scores = (Q*scale) K^T  (+ key mask) ----------------
    for (int ct = wave; ct < Sc / 16; ct += 4) {
        const int n0 = ct << 4;
        // B layout (16-bit, 32x16): lanes 0-15 hold K {0..7, 16..23}; lanes 16-31 hold K {8..15, 24..31}
        v16bf bk[2];
        const float* krow = Kh + (size_t)(n0 + ln) * Dc;
        #pragma unroll
        for (int c = 0; c < 2; ++c) {
            const int kb = c * 32 + (lh ? 8 : 0);
            float t[16];
            *(float4*)(t + 0)  = *(const float4*)(krow + kb + 0);
            *(float4*)(t + 4)  = *(const float4*)(krow + kb + 4);
            *(float4*)(t + 8)  = *(const float4*)(krow + kb + 16);
            *(float4*)(t + 12) = *(const float4*)(krow + kb + 20);
            v16bf f;
            #pragma unroll
            for (int e = 0; e < 16; ++e) f[e] = (bf16)t[e];
            bk[c] = f;
        }

        v8f acc = {};
        acc = WMMA_BF16(aq[0], bk[0], acc);
        acc = WMMA_BF16(aq[1], bk[1], acc);

        const int mv = Mb[n0 + ln];          // mask for this lane's key column
        float* dst = sc + (size_t)(8 * lh) * LDSW + (n0 + ln);
        #pragma unroll
        for (int i = 0; i < 8; ++i) {        // row = i + 8*lh, col = n0 + ln
            float s = (mv == 0) ? NEGV : acc[i];
            dst[(size_t)i * LDSW] = s;
        }
    }
    __syncthreads();

    // ---------------- Phase B: exact softmax over 2048 keys per row ----------------
    {
        const int row = tid >> 3;            // 0..15
        const int seg = tid & 7;             // 0..7
        float* srow = sc + (size_t)row * LDSW;

        float m = -3.4e38f;
        for (int i = seg; i < Sc; i += 8) m = fmaxf(m, srow[i]);
        red[row * 8 + seg] = m;
        __syncthreads();
        if (seg == 0) {
            float mm = red[row * 8];
            #pragma unroll
            for (int j = 1; j < 8; ++j) mm = fmaxf(mm, red[row * 8 + j]);
            mxA[row] = mm;
        }
        __syncthreads();

        const float mm = mxA[row];
        float s = 0.f;
        for (int i = seg; i < Sc; i += 8) {
            float e = __expf(srow[i] - mm);
            srow[i] = e;
            s += e;
        }
        red[row * 8 + seg] = s;
        __syncthreads();
        if (seg == 0) {
            float ss = 0.f;
            #pragma unroll
            for (int j = 0; j < 8; ++j) ss += red[row * 8 + j];
            invA[row] = 1.f / ss;
        }
        __syncthreads();

        // Normalize in LDS and stream attn_weights to HBM fully coalesced.
        // Non-temporal: 512 MB write-once stream must not evict K/V working set from L2.
        const size_t pbase = ((size_t)bh * Sc + q0) * Sc;
        for (int idx = tid; idx < 16 * Sc; idx += 128) {
            const int r = idx >> 11;
            const int c = idx & (Sc - 1);
            const float p = sc[(size_t)r * LDSW + c] * invA[r];
            sc[(size_t)r * LDSW + c] = p;
            __builtin_nontemporal_store(p, P + pbase + (size_t)r * Sc + c);
        }
        __syncthreads();
    }

    // ---------------- Phase C: out = P @ V (each wave owns a 16-wide dim tile) ----------------
    {
        const int w16 = wave << 4;
        v8f oacc = {};
        for (int kc = 0; kc < Sc / 32; ++kc) {
            // A fragment from normalized probabilities in LDS
            v16bf ap;
            {
                const float* prow = sc + (size_t)ln * LDSW + kc * 32 + (lh ? 8 : 0);
                float t[16];
                *(float4*)(t + 0)  = *(const float4*)(prow + 0);
                *(float4*)(t + 4)  = *(const float4*)(prow + 4);
                *(float4*)(t + 8)  = *(const float4*)(prow + 16);
                *(float4*)(t + 12) = *(const float4*)(prow + 20);
                #pragma unroll
                for (int e = 0; e < 16; ++e) ap[e] = (bf16)t[e];
            }
            // B fragment from V: B(kd, n) = V[kc*32 + kd][w16 + n]
            v16bf bv;
            {
                const float* vp = Vh + (size_t)(kc * 32 + (lh ? 8 : 0)) * Dc + (w16 + ln);
                float t[16];
                #pragma unroll
                for (int e = 0; e < 16; ++e) t[e] = vp[(size_t)(e < 8 ? e : e + 8) * Dc];
                #pragma unroll
                for (int e = 0; e < 16; ++e) bv[e] = (bf16)t[e];
            }
            oacc = WMMA_BF16(ap, bv, oacc);
        }
        // store 16x16 output tile: row = i + 8*lh, col = w16 + ln (non-temporal, write-once)
        #pragma unroll
        for (int i = 0; i < 8; ++i) {
            __builtin_nontemporal_store(
                oacc[i], O + ((size_t)bh * Sc + q0 + 8 * lh + i) * Dc + w16 + ln);
        }
    }
}

extern "C" void kernel_launch(void* const* d_in, const int* in_sizes, int n_in,
                              void* d_out, int out_size, void* d_ws, size_t ws_size,
                              hipStream_t stream) {
    const float* q  = (const float*)d_in[0];
    const float* k  = (const float*)d_in[1];
    const float* v  = (const float*)d_in[2];
    const int*   am = (const int*)d_in[3];

    float* out   = (float*)d_out;                              // (B,H,S,D) first
    float* attnw = out + (size_t)Bc * Hc * Sc * Dc;            // then (B,H,S,S)

    const size_t smemBytes = (size_t)(16 * LDSW + 128 + 16 + 16) * sizeof(float); // ~132 KB < 320 KB/WGP
    const dim3 grid(Bc * Hc * (Sc / 16));                      // 4096 workgroups
    sdpa_fwd<<<grid, 128, smemBytes, stream>>>(q, k, v, am, out, attnw);
}
